// CtcAdaptor_78872779424114
// MI455X (gfx1250) — matrix operand order repaired and checked by reading, not compiled, not run on hardware
//
#include <hip/hip_runtime.h>
#include <hip/hip_bf16.h>
#include <math.h>
#include <stdint.h>

// Problem constants (from reference: T=256, B=8, D=512, V=10000, UP=2)
#define Tsrc  256
#define Bn    8
#define Dm    512
#define Vv    10000
#define TUP   512              // T * UP
#define Mrows (Bn * TUP)       // 4096 rows through both GEMMs

typedef __attribute__((ext_vector_type(16))) __bf16 v16bf;
typedef __attribute__((ext_vector_type(8)))  __bf16 v8bf;
typedef __attribute__((ext_vector_type(8)))  float  v8f;
typedef __attribute__((ext_vector_type(4)))  int    v4i;

typedef __attribute__((address_space(1))) v4i gv4i;   // global v4i
typedef __attribute__((address_space(3))) v4i lv4i;   // LDS v4i

static __device__ __forceinline__ unsigned short f2bf(float f) {
  unsigned u = __builtin_bit_cast(unsigned, f);
  u += 0x7FFFu + ((u >> 16) & 1u);          // round-to-nearest-even
  return (unsigned short)(u >> 16);
}

// 16-byte global -> LDS copy through the CDNA5 async DMA path when available.
static __device__ __forceinline__ void async_cp16(const void* g, void* l) {
#if __has_builtin(__builtin_amdgcn_global_load_async_to_lds_b128)
  __builtin_amdgcn_global_load_async_to_lds_b128(
      (gv4i*)(unsigned long long)(size_t)g,
      (lv4i*)(unsigned)(size_t)l, 0, 0);
#else
  *(uint4*)l = *(const uint4*)g;
#endif
}

static __device__ __forceinline__ void async_wait0() {
#if __has_builtin(__builtin_amdgcn_global_load_async_to_lds_b128)
# if __has_builtin(__builtin_amdgcn_s_wait_asynccnt)
  __builtin_amdgcn_s_wait_asynccnt(0);
# else
  asm volatile("s_wait_asynccnt 0x0" ::: "memory");
# endif
#endif
}

// Assemble A fragment: lane holds row m, K = hi*16..hi*16+15 (contiguous)
static __device__ __forceinline__ v16bf load_a_frag(const unsigned short* rowp, int k0, int hi) {
  return *(const v16bf*)(rowp + k0 + hi * 16);
}

// ---------------- upsample bookkeeping ----------------
__global__ void k_src_lens(const int* __restrict__ mask, int* __restrict__ lens) {
  __shared__ int cnt;
  const int b = blockIdx.x, t = threadIdx.x;
  if (t == 0) cnt = 0;
  __syncthreads();
  if (mask[b * Tsrc + t] == 0) atomicAdd(&cnt, 1);
  __syncthreads();
  if (t == 0) lens[b] = cnt;
}

__global__ void k_map(const int* __restrict__ lens, int* __restrict__ srcIdx,
                      int* __restrict__ padF) {
  const int r = blockIdx.x * blockDim.x + threadIdx.x;
  if (r >= Mrows) return;
  const int b = r >> 9, t = r & (TUP - 1);
  const int sl = lens[b];
  const int ul = sl * 2;
  const int pad = (t >= ul) ? 1 : 0;
  int src = 0;
  if (!pad && sl > 0) {
    const float ratio = (float)ul / (float)sl;
    src = (int)((float)t / ratio);
    if (src >= Tsrc) src = Tsrc - 1;
  }
  srcIdx[r] = src;
  padF[r]   = pad;
}

// ---------------- f32 -> bf16 bulk convert (for E) ----------------
__global__ void k_f32_to_bf16(const float* __restrict__ src,
                              unsigned short* __restrict__ dst, int n) {
  const int i = blockIdx.x * blockDim.x + threadIdx.x;
  if (i < n) dst[i] = f2bf(src[i]);
}

// ---------------- W: f32 [k][n] -> bf16 transposed [n][k] ----------------
__global__ void k_w_t_bf16(const float* __restrict__ W, unsigned short* __restrict__ Wt) {
  const int i = blockIdx.x * blockDim.x + threadIdx.x;   // over Dm*Dm
  const int k = i >> 9, n = i & (Dm - 1);
  Wt[(size_t)n * Dm + k] = f2bf(W[(size_t)k * Dm + n]);
}

// ---------------- GEMM 1: h = gather(x) @ W_proj  (bf16 WMMA, f32 acc) ----
// 128 threads = 4 waves; block tile 16(M) x 128(N); wave computes 16x32.
// B tile double-buffered and staged by async global->LDS DMA.
__global__ void __launch_bounds__(128)
k_proj_gemm(const float* __restrict__ x, const unsigned short* __restrict__ Wt,
            const int* __restrict__ srcIdx, float* __restrict__ h) {
  __shared__ unsigned short sA[16][Dm];        // full K row-tile, 16 KB
  __shared__ unsigned short sB[2][128][32];    // transposed [n][k], 2 x 8 KB
  const int tid  = threadIdx.x;
  const int lane = tid & 31, wv = tid >> 5;
  const int m0 = blockIdx.x * 16;
  const int n0 = blockIdx.y * 128;
  {                                            // stage A once (gather + convert)
    const int row   = tid >> 3;                // 0..15
    const int kbase = (tid & 7) * 64;          // 0..448
    const int r = m0 + row;
    const int b = r >> 9;
    const float* xr = &x[(size_t)(srcIdx[r] * Bn + b) * Dm];
    for (int k = 0; k < 64; ++k) sA[row][kbase + k] = f2bf(xr[kbase + k]);
  }
  // prime B buffer 0 (thread owns column n0+tid; 4 x 16B async copies)
  {
    const unsigned short* src = &Wt[(size_t)(n0 + tid) * Dm];
    #pragma unroll
    for (int i = 0; i < 4; ++i) async_cp16(src + i * 8, &sB[0][tid][i * 8]);
  }
  async_wait0();
  __syncthreads();
  v8f acc0 = {}, acc1 = {};
  const int mrow = lane & 15, hi = lane >> 4;
  for (int k0 = 0; k0 < Dm; k0 += 32) {
    const int cur = (k0 >> 5) & 1;
    if (k0 + 32 < Dm) {                        // async-stage next B tile
      const unsigned short* src = &Wt[(size_t)(n0 + tid) * Dm + k0 + 32];
      #pragma unroll
      for (int i = 0; i < 4; ++i) async_cp16(src + i * 8, &sB[cur ^ 1][tid][i * 8]);
    }
    const v16bf av  = load_a_frag(&sA[mrow][0], k0, hi);
    const v16bf bv0 = *(const v16bf*)&sB[cur][wv * 32 + mrow][hi * 16];
    const v16bf bv1 = *(const v16bf*)&sB[cur][wv * 32 + 16 + mrow][hi * 16];
    acc0 = __builtin_amdgcn_wmma_f32_16x16x32_bf16(false, av, false, bv0,
                                                   (short)0, acc0, false, false);
    acc1 = __builtin_amdgcn_wmma_f32_16x16x32_bf16(false, av, false, bv1,
                                                   (short)0, acc1, false, false);
    async_wait0();
    __syncthreads();
  }
  #pragma unroll
  for (int i = 0; i < 8; ++i) {                // C: M = i + 8*hi, N = lane%16
    const int row = i + 8 * hi;
    const int col = lane & 15;
    h[(size_t)(m0 + row) * Dm + n0 + wv * 32 + col]      = acc0[i];
    h[(size_t)(m0 + row) * Dm + n0 + wv * 32 + 16 + col] = acc1[i];
  }
}

// ---------------- bias + sinusoidal positions (writes f32 + bf16 h) -------
__global__ void k_bias_pos(float* __restrict__ h, unsigned short* __restrict__ hbf,
                           const float* __restrict__ bproj, const int* __restrict__ padF) {
  const int r = blockIdx.x;           // 0..4095
  const int d = threadIdx.x;          // 0..511
  const int t = r & (TUP - 1);
  const int pos = padF[r] ? 1 : (t + 2);   // fairseq make_positions (+PAD)
  float pe = 0.f;
  if (pos != 1) {                          // tab[PAD] == 0
    const int half = Dm / 2;
    const int i = (d < half) ? d : d - half;
    const float freq = __expf(-__logf(10000.f) * (float)i / (float)(half - 1));
    const float ang  = (float)pos * freq;
    pe = (d < half) ? __sinf(ang) : __cosf(ang);
  }
  const float v = h[(size_t)r * Dm + d] + bproj[d] + pe;
  h[(size_t)r * Dm + d]   = v;
  hbf[(size_t)r * Dm + d] = f2bf(v);
}

// ---------------- GEMM 2: logits = h @ E^T fused with online softmax ------
// Per 16-row tile, sweep V in 128-col chunks; E tile double-buffered via
// async global->LDS DMA; 2 WMMAs per wave per k-step; online
// (max, argmax, sumexp) per row.  probs = exp(lp[argmax]) = 1 / sumexp.
__global__ void __launch_bounds__(128)
k_ctc_logits(const unsigned short* __restrict__ hbf, const unsigned short* __restrict__ Ebf,
             const int* __restrict__ padF, int* __restrict__ pred,
             float* __restrict__ probs) {
  __shared__ unsigned short sA[16][Dm];        // 16 KB, staged once
  __shared__ unsigned short sB[2][128][32];    // transposed [n][k], 2 x 8 KB
  __shared__ float sL[16][128];                // chunk logits, 8 KB
  const int tid  = threadIdx.x;
  const int lane = tid & 31, wv = tid >> 5;
  const int m0 = blockIdx.x * 16;
  {                                            // stage A row-tile once (vectorized)
    const int row   = tid >> 3;
    const int kbase = (tid & 7) * 64;
    const uint4* src = (const uint4*)&hbf[(size_t)(m0 + row) * Dm + kbase];
    uint4* dst = (uint4*)&sA[row][kbase];
    #pragma unroll
    for (int i = 0; i < 8; ++i) dst[i] = src[i];
  }
  const int mrow = lane & 15, hi = lane >> 4;
  float runMax = -__builtin_inff();
  float runSum = 0.f;
  int   runArg = 0;
  const int NC = (Vv + 127) / 128;
  for (int chunk = 0; chunk < NC; ++chunk) {
    const int v0 = chunk * 128;
    // column owned by this thread; clamp OOB to stay in-bounds (cols >= V are
    // computed but never read: the scan below masks with nv).
    int v = v0 + tid; if (v >= Vv) v = Vv - 1;
    {                                          // prime B buffer 0 for k0 = 0
      const unsigned short* src = &Ebf[(size_t)v * Dm];
      #pragma unroll
      for (int i = 0; i < 4; ++i) async_cp16(src + i * 8, &sB[0][tid][i * 8]);
    }
    if (v0 + 128 + tid < Vv)                   // -> global_prefetch_b8 (next chunk)
      __builtin_prefetch(&Ebf[(size_t)(v0 + 128 + tid) * Dm], 0, 1);
    async_wait0();
    __syncthreads();
    v8f acc0 = {}, acc1 = {};
    for (int k0 = 0; k0 < Dm; k0 += 32) {
      const int cur = (k0 >> 5) & 1;
      if (k0 + 32 < Dm) {                      // async-stage next E tile
        const unsigned short* src = &Ebf[(size_t)v * Dm + k0 + 32];
        #pragma unroll
        for (int i = 0; i < 4; ++i) async_cp16(src + i * 8, &sB[cur ^ 1][tid][i * 8]);
      }
      const v16bf av  = load_a_frag(&sA[mrow][0], k0, hi);
      const v16bf bv0 = *(const v16bf*)&sB[cur][wv * 32 + mrow][hi * 16];
      const v16bf bv1 = *(const v16bf*)&sB[cur][wv * 32 + 16 + mrow][hi * 16];
      acc0 = __builtin_amdgcn_wmma_f32_16x16x32_bf16(false, av, false, bv0,
                                                     (short)0, acc0, false, false);
      acc1 = __builtin_amdgcn_wmma_f32_16x16x32_bf16(false, av, false, bv1,
                                                     (short)0, acc1, false, false);
      async_wait0();
      __syncthreads();
    }
    #pragma unroll
    for (int i = 0; i < 8; ++i) {
      const int row = i + 8 * hi;
      const int col = lane & 15;
      sL[row][wv * 32 + col]      = acc0[i];
      sL[row][wv * 32 + 16 + col] = acc1[i];
    }
    __syncthreads();
    if (tid < 16) {                            // thread t owns row m0+t
      int nv = Vv - v0; if (nv > 128) nv = 128;
      for (int c = 0; c < nv; ++c) {
        const float val = sL[tid][c];
        if (val > runMax) {
          runSum = runSum * __expf(runMax - val) + 1.f;
          runMax = val;
          runArg = v0 + c;
        } else {
          runSum += __expf(val - runMax);
        }
      }
    }
    __syncthreads();
  }
  if (tid < 16) {
    const int r = m0 + tid;
    pred[r]  = padF[r] ? 0 : runArg;
    probs[r] = 1.f / runSum;                   // exp(max - lse)
  }
}

// ---------------- per-batch segment scan: index, segment softmax weights --
__global__ void k_scan(const int* __restrict__ pred, const float* __restrict__ probs,
                       int* __restrict__ segIdx, float* __restrict__ segW,
                       int* __restrict__ tgtLens, float* __restrict__ out_pad_f,
                       float* __restrict__ out_pred_f) {
  __shared__ int   sIdx[TUP];
  __shared__ float sMax[TUP + 1];
  __shared__ float sSum[TUP + 1];
  __shared__ int   sTgt;
  const int b = blockIdx.x, t = threadIdx.x;
  for (int i = t; i <= TUP; i += blockDim.x) { sMax[i] = -__builtin_inff(); sSum[i] = 0.f; }
  __syncthreads();
  if (t == 0) {
    int run = 0, prev = -1, tgt = 0;
    for (int j = 0; j < TUP; ++j) {
      const int p = pred[b * TUP + j];
      const int msk = (p == 0);
      const int d = (!msk && p != prev) ? 1 : 0;
      run += msk ? 0 : d;
      const int idx = msk ? 0 : run;
      sIdx[j] = idx;
      if (idx > tgt) tgt = idx;
      prev = p;
    }
    sTgt = tgt;
    for (int j = 0; j < TUP; ++j) {                  // segment max
      const int idx = sIdx[j];
      if (idx > 0) { const float pv = probs[b * TUP + j]; if (pv > sMax[idx]) sMax[idx] = pv; }
    }
    for (int j = 0; j < TUP; ++j) {                  // segment sum-exp
      const int idx = sIdx[j];
      if (idx > 0) sSum[idx] += __expf(probs[b * TUP + j] - sMax[idx]);
    }
  }
  __syncthreads();
  {
    const int idx = sIdx[t];
    float w = 0.f;
    if (idx > 0) w = __expf(probs[b * TUP + t] - sMax[idx]) / sSum[idx];
    segIdx[b * TUP + t]     = idx;
    segW[b * TUP + t]       = w;
    out_pad_f[b * TUP + t]  = (t >= sTgt) ? 1.f : 0.f;
    out_pred_f[b * TUP + t] = (float)pred[b * TUP + t];
    if (t == 0) tgtLens[b] = sTgt;
  }
}

// ---------------- merge: weighted segment sums, output [Tup, B, D] --------
__global__ void k_merge(const float* __restrict__ h, const int* __restrict__ segIdx,
                        const float* __restrict__ segW, float* __restrict__ out_merge) {
  const int blk = blockIdx.x;               // 0..4095 = (b, s)
  const int b = blk >> 9;
  const int s = blk & (TUP - 1);            // segment id s+1 -> output row s
  const int d = threadIdx.x;
  float acc = 0.f;
  for (int t = 0; t < TUP; ++t) {
    if (segIdx[b * TUP + t] == s + 1) {
      acc += h[(size_t)(b * TUP + t) * Dm + d] * segW[b * TUP + t];
    }
  }
  out_merge[(size_t)(s * Bn + b) * Dm + d] = acc;
}

extern "C" void kernel_launch(void* const* d_in, const int* in_sizes, int n_in,
                              void* d_out, int out_size, void* d_ws, size_t ws_size,
                              hipStream_t stream) {
  const float* x     = (const float*)d_in[0];   // [T,B,D]
  const int*   mask  = (const int*)d_in[1];     // [B,T]
  const float* W     = (const float*)d_in[2];   // [D,D]
  const float* bproj = (const float*)d_in[3];   // [D]
  const float* E     = (const float*)d_in[4];   // [V,D]

  char* ws = (char*)d_ws;
  size_t off = 0;
  float*          h      = (float*)(ws + off);          off += (size_t)Mrows * Dm * 4; // 8 MB
  unsigned short* hbf    = (unsigned short*)(ws + off); off += (size_t)Mrows * Dm * 2; // 4 MB
  unsigned short* Ebf    = (unsigned short*)(ws + off); off += (size_t)Vv * Dm * 2;    // 10 MB
  unsigned short* Wt     = (unsigned short*)(ws + off); off += (size_t)Dm * Dm * 2;    // 0.5 MB
  int*            srcIdx = (int*)(ws + off);            off += (size_t)Mrows * 4;
  int*            padF   = (int*)(ws + off);            off += (size_t)Mrows * 4;
  int*            pred   = (int*)(ws + off);            off += (size_t)Mrows * 4;
  float*          probs  = (float*)(ws + off);          off += (size_t)Mrows * 4;
  int*            segIdx = (int*)(ws + off);            off += (size_t)Mrows * 4;
  float*          segW   = (float*)(ws + off);          off += (size_t)Mrows * 4;
  int*            lens   = (int*)(ws + off);            off += 64;
  int*            tgt    = (int*)(ws + off);            off += 64;

  float* out       = (float*)d_out;
  float* out_merge = out;                                  // [Tup,B,D]
  float* out_pad   = out + (size_t)TUP * Bn * Dm;          // [B,Tup]
  float* out_pred  = out_pad + (size_t)Bn * TUP;           // [B,Tup]

  const int nE = Vv * Dm;
  const int nW = Dm * Dm;
  k_src_lens   <<<dim3(Bn),                dim3(Tsrc), 0, stream>>>(mask, lens);
  k_map        <<<dim3(Mrows / 256),       dim3(256),  0, stream>>>(lens, srcIdx, padF);
  k_f32_to_bf16<<<dim3((nE + 255) / 256),  dim3(256),  0, stream>>>(E, Ebf, nE);
  k_w_t_bf16   <<<dim3(nW / 256),          dim3(256),  0, stream>>>(W, Wt);
  k_proj_gemm  <<<dim3(Mrows / 16, Dm/128),dim3(128),  0, stream>>>(x, Wt, srcIdx, h);
  k_bias_pos   <<<dim3(Mrows),             dim3(Dm),   0, stream>>>(h, hbf, bproj, padF);
  k_ctc_logits <<<dim3(Mrows / 16),        dim3(128),  0, stream>>>(hbf, Ebf, padF, pred, probs);
  k_scan       <<<dim3(Bn),                dim3(TUP),  0, stream>>>(pred, probs, segIdx, segW,
                                                                    tgt, out_pad, out_pred);
  k_merge      <<<dim3(Mrows),             dim3(Dm),   0, stream>>>(h, segIdx, segW, out_merge);
}
